// MultiHeadAttention_21638045237284
// MI455X (gfx1250) — compile-verified, matrix operand order checked by reading, not run on hardware
//
#include <hip/hip_runtime.h>

#define BB 4
#define TT 2048
#define CC 1024
#define HH 16
#define HS 64

typedef __attribute__((ext_vector_type(16))) __bf16 v16bf;
typedef __attribute__((ext_vector_type(8)))  __bf16 v8bf;
typedef __attribute__((ext_vector_type(8)))  float  v8f;

// ---------------------------------------------------------------------------
// Async global->LDS copy (CDNA5 GLOBAL_LOAD_ASYNC_TO_LDS_B128, ASYNCcnt).
// Builtin signature (from hipcc diagnostic): (v4i32 global*, v4i32 lds*, Ii, Ii)
// ---------------------------------------------------------------------------
#if defined(__has_builtin)
#if __has_builtin(__builtin_amdgcn_global_load_async_to_lds_b128)
#define HAVE_ASYNC_LDS 1
#endif
#endif
#ifndef HAVE_ASYNC_LDS
#define HAVE_ASYNC_LDS 0
#endif

typedef int v4i_vs __attribute__((vector_size(16)));
typedef __attribute__((address_space(1))) v4i_vs* as1_v4i_ptr;
typedef __attribute__((address_space(3))) v4i_vs* as3_v4i_ptr;

__device__ __forceinline__ void async_copy_b128(void* lds_dst, const void* gsrc) {
#if HAVE_ASYNC_LDS
  __builtin_amdgcn_global_load_async_to_lds_b128(
      (as1_v4i_ptr)(void*)gsrc, (as3_v4i_ptr)lds_dst, 0, 0);
#else
  *(v8bf*)lds_dst = *(const v8bf*)gsrc;
#endif
}

__device__ __forceinline__ void async_wait() {
#if HAVE_ASYNC_LDS
  asm volatile("s_wait_asynccnt 0" ::: "memory");
#endif
}

// ---------------------------------------------------------------------------
// A/B fragment load for v_wmma_f32_16x16x32_bf16 from a row-major matrix.
// lane L holds row (L&15); half = L>>4 selects K groups:
//   elements 0..7  = row[8*half .. +7], elements 8..15 = row[16+8*half .. +7]
// ---------------------------------------------------------------------------
__device__ __forceinline__ v16bf frag_load(const __bf16* rowp, int half) {
  v8bf lo = *(const v8bf*)(rowp + 8 * half);
  v8bf hi = *(const v8bf*)(rowp + 16 + 8 * half);
  return __builtin_shufflevector(lo, hi, 0,1,2,3,4,5,6,7,8,9,10,11,12,13,14,15);
}

__device__ __forceinline__ v8f wmma_bf16(v16bf a, v16bf b, v8f c) {
  return __builtin_amdgcn_wmma_f32_16x16x32_bf16(false, a, false, b, (short)0, c,
                                                 false, false);
}

// ---------------------------------------------------------------------------
// Kernel 0: f32 -> bf16 bulk convert (grid-stride)
// ---------------------------------------------------------------------------
__global__ void f32_to_bf16(const float* __restrict__ src,
                            __bf16* __restrict__ dst, size_t n) {
  size_t i = (size_t)blockIdx.x * blockDim.x + threadIdx.x;
  size_t stride = (size_t)gridDim.x * blockDim.x;
  for (; i < n; i += stride) dst[i] = (__bf16)src[i];
}

// ---------------------------------------------------------------------------
// Kernel 1: fused QKV projection.
// grid = (B*T/128, H, 3), block = 128 (4 waves).
// Block tile 128x64; each wave owns a 32x64 strip (2 A-frags x 4 col-tiles
// -> 8 WMMA per 32-wide K chunk).
// ---------------------------------------------------------------------------
__global__ __launch_bounds__(128)
void qkv_gemm(const __bf16* __restrict__ xb,
              const __bf16* __restrict__ wq,
              const __bf16* __restrict__ wk,
              const __bf16* __restrict__ wv,
              __bf16* __restrict__ qo,
              __bf16* __restrict__ ko,
              __bf16* __restrict__ vo) {
  __shared__ __bf16 Xt[128][32];  // 8 KB
  __shared__ __bf16 Wt[64][32];   // 4 KB, transposed: [out-col][k]
  const int mt = blockIdx.x, h = blockIdx.y, which = blockIdx.z;
  const __bf16* W = (which == 0 ? wq : which == 1 ? wk : wv) + (size_t)h * CC * HS;
  __bf16* O = (which == 0 ? qo : which == 1 ? ko : vo);
  const int tid = threadIdx.x;
  const int wvi = tid >> 5, lane = tid & 31;
  const int col = lane & 15, half = lane >> 4;
  const int m0 = mt * 128;
  const int b = m0 / TT;

  v8f acc[2][4] = {};
  for (int kc = 0; kc < CC; kc += 32) {
    for (int c = tid; c < 512; c += 128) {           // X tile: 4096 elems / 8
      int rr = c >> 2, k0 = (c & 3) * 8;
      async_copy_b128(&Xt[rr][k0], xb + (size_t)(m0 + rr) * CC + kc + k0);
    }
    for (int c = tid; c < 256; c += 128) {           // W tile (transpose in LDS)
      int kk = c >> 3, n0 = (c & 7) * 8;
      v8bf w = *(const v8bf*)(W + (size_t)(kc + kk) * HS + n0);
#pragma unroll
      for (int j = 0; j < 8; ++j) Wt[n0 + j][kk] = w[j];
    }
    if (kc + 32 < CC) {                              // prefetch next chunk
      __builtin_prefetch(xb + (size_t)(m0 + (tid & 127)) * CC + kc + 32, 0, 1);
      __builtin_prefetch(W + (size_t)(kc + 32 + (tid & 31)) * HS, 0, 1);
    }
    async_wait();
    __syncthreads();
    v16bf a0 = frag_load(&Xt[wvi * 32 + col][0], half);
    v16bf a1 = frag_load(&Xt[wvi * 32 + 16 + col][0], half);
#pragma unroll
    for (int n = 0; n < 4; ++n) {
      v16bf bf = frag_load(&Wt[n * 16 + col][0], half);
      acc[0][n] = wmma_bf16(a0, bf, acc[0][n]);
      acc[1][n] = wmma_bf16(a1, bf, acc[1][n]);
    }
    __syncthreads();
  }
#pragma unroll
  for (int mi = 0; mi < 2; ++mi)
#pragma unroll
    for (int n = 0; n < 4; ++n)
#pragma unroll
      for (int r = 0; r < 8; ++r) {
        int g = m0 + wvi * 32 + mi * 16 + r + 8 * half;
        int t = g - b * TT;
        O[(((size_t)b * HH + h) * TT + t) * HS + n * 16 + col] =
            (__bf16)acc[mi][n][r];
      }
}

// ---------------------------------------------------------------------------
// Kernel 2: causal flash attention, one (b,h) x 64-query block per workgroup.
// grid = (T/64, B*H), block = 128 (4 waves; wave = 16 query rows, full HS=64).
// ---------------------------------------------------------------------------
__global__ __launch_bounds__(128)
void flash_attn(const __bf16* __restrict__ qg, const __bf16* __restrict__ kg,
                const __bf16* __restrict__ vg, __bf16* __restrict__ att) {
  __shared__ __bf16 Kt[64][64];       // key-major
  __shared__ __bf16 VT[64][64];       // [hs-dim][key] (transposed)
  __shared__ __bf16 Pb[4][16][64];    // per-wave P staging (C/D -> A relayout)
  const int qb = blockIdx.x, bh = blockIdx.y;
  const int b = bh / HH, h = bh % HH;
  const size_t base = (size_t)bh * TT * HS;
  const __bf16* Q = qg + base;
  const __bf16* K = kg + base;
  const __bf16* V = vg + base;
  const int tid = threadIdx.x, wvi = tid >> 5, lane = tid & 31;
  const int col = lane & 15, half = lane >> 4;

  const int qrow = qb * 64 + wvi * 16 + col;          // A-fragment row
  v16bf qf0 = frag_load(Q + (size_t)qrow * HS, half);
  v16bf qf1 = frag_load(Q + (size_t)qrow * HS + 32, half);

  v8f o[4] = {};
  float m_run[8], l_run[8];
#pragma unroll
  for (int r = 0; r < 8; ++r) { m_run[r] = -1e30f; l_run[r] = 0.0f; }

  for (int kblk = 0; kblk <= qb; ++kblk) {
    // stage K tile (async, no transpose) and V^T tile (scatter transpose)
    for (int c = tid; c < 512; c += 128) {
      int s = c >> 3, d0 = (c & 7) * 8;
      async_copy_b128(&Kt[s][d0], K + (size_t)(kblk * 64 + s) * HS + d0);
      v8bf vv = *(const v8bf*)(V + (size_t)(kblk * 64 + s) * HS + d0);
#pragma unroll
      for (int j = 0; j < 8; ++j) VT[d0 + j][s] = vv[j];
    }
    if (kblk + 1 <= qb) {                             // prefetch next key block
      __builtin_prefetch(K + (size_t)((kblk + 1) * 64 + (tid >> 1)) * HS, 0, 1);
      __builtin_prefetch(V + (size_t)((kblk + 1) * 64 + (tid >> 1)) * HS, 0, 1);
    }
    async_wait();
    __syncthreads();

    // S = Q * K^T  (B columns = key rows of Kt -> contiguous loads)
    v8f sa[4];
#pragma unroll
    for (int n = 0; n < 4; ++n) {
      const __bf16* kp = &Kt[n * 16 + col][0];
      v16bf b0 = frag_load(kp, half);
      v16bf b1 = frag_load(kp + 32, half);
      v8f c = {};
      c = wmma_bf16(qf0, b0, c);
      c = wmma_bf16(qf1, b1, c);
      sa[n] = c;
    }

    // causal mask: only the diagonal block can violate col<=row
    if (kblk == qb) {
#pragma unroll
      for (int n = 0; n < 4; ++n)
#pragma unroll
        for (int r = 0; r < 8; ++r) {
          int rowg = qb * 64 + wvi * 16 + r + 8 * half;
          int colg = kblk * 64 + n * 16 + col;
          if (colg > rowg) sa[n][r] = -1e30f;
        }
    }

    // online softmax: row r+8*half lives on the 16 lanes of this half
#pragma unroll
    for (int r = 0; r < 8; ++r) {
      float mr = fmaxf(fmaxf(sa[0][r], sa[1][r]), fmaxf(sa[2][r], sa[3][r]));
#pragma unroll
      for (int off = 8; off > 0; off >>= 1) mr = fmaxf(mr, __shfl_xor(mr, off, 32));
      float mnew = fmaxf(m_run[r], mr);
      float rescale = __expf(m_run[r] - mnew);
      float rs = 0.0f;
#pragma unroll
      for (int n = 0; n < 4; ++n) {
        float p = __expf(sa[n][r] - mnew);
        rs += p;
        Pb[wvi][r + 8 * half][n * 16 + col] = (__bf16)p;
      }
#pragma unroll
      for (int off = 8; off > 0; off >>= 1) rs += __shfl_xor(rs, off, 32);
      l_run[r] = l_run[r] * rescale + rs;
      m_run[r] = mnew;
#pragma unroll
      for (int dn = 0; dn < 4; ++dn) o[dn][r] *= rescale;
    }

    // O += P * V   (A = P from LDS staging, B columns = rows of V^T)
    const __bf16* pp = &Pb[wvi][col][0];
    v16bf p0 = frag_load(pp, half);
    v16bf p1 = frag_load(pp + 32, half);
#pragma unroll
    for (int dn = 0; dn < 4; ++dn) {
      const __bf16* vp = &VT[dn * 16 + col][0];
      o[dn] = wmma_bf16(p0, frag_load(vp, half), o[dn]);
      o[dn] = wmma_bf16(p1, frag_load(vp + 32, half), o[dn]);
    }
    __syncthreads();
  }

  // normalize and write concat-heads layout (B,T,C), bf16 for the projection
#pragma unroll
  for (int dn = 0; dn < 4; ++dn)
#pragma unroll
    for (int r = 0; r < 8; ++r) {
      int t = qb * 64 + wvi * 16 + r + 8 * half;
      int d = dn * 16 + col;
      att[((size_t)b * TT + t) * CC + h * HS + d] = (__bf16)(o[dn][r] / l_run[r]);
    }
}

// ---------------------------------------------------------------------------
// Kernel 3: output projection  out = att(8192x1024) * proj_w + proj_b (f32 out)
// grid = (B*T/128, C/64), block = 128.  Same 32x64-per-wave register tiling.
// ---------------------------------------------------------------------------
__global__ __launch_bounds__(128)
void proj_gemm(const __bf16* __restrict__ att, const __bf16* __restrict__ pw,
               const float* __restrict__ pbias, float* __restrict__ out) {
  __shared__ __bf16 Xt[128][32];
  __shared__ __bf16 Wt[64][32];
  const int mt = blockIdx.x, nt = blockIdx.y;
  const int tid = threadIdx.x, wvi = tid >> 5, lane = tid & 31;
  const int col = lane & 15, half = lane >> 4;
  const int m0 = mt * 128, n0g = nt * 64;

  v8f acc[2][4] = {};
  for (int kc = 0; kc < CC; kc += 32) {
    for (int c = tid; c < 512; c += 128) {
      int rr = c >> 2, k0 = (c & 3) * 8;
      async_copy_b128(&Xt[rr][k0], att + (size_t)(m0 + rr) * CC + kc + k0);
    }
    for (int c = tid; c < 256; c += 128) {
      int kk = c >> 3, nn = (c & 7) * 8;
      v8bf w = *(const v8bf*)(pw + (size_t)(kc + kk) * CC + n0g + nn);
#pragma unroll
      for (int j = 0; j < 8; ++j) Wt[nn + j][kk] = w[j];
    }
    if (kc + 32 < CC) {
      __builtin_prefetch(att + (size_t)(m0 + (tid & 127)) * CC + kc + 32, 0, 1);
      __builtin_prefetch(pw + (size_t)(kc + 32 + (tid & 31)) * CC + n0g, 0, 1);
    }
    async_wait();
    __syncthreads();
    v16bf a0 = frag_load(&Xt[wvi * 32 + col][0], half);
    v16bf a1 = frag_load(&Xt[wvi * 32 + 16 + col][0], half);
#pragma unroll
    for (int n = 0; n < 4; ++n) {
      v16bf bf = frag_load(&Wt[n * 16 + col][0], half);
      acc[0][n] = wmma_bf16(a0, bf, acc[0][n]);
      acc[1][n] = wmma_bf16(a1, bf, acc[1][n]);
    }
    __syncthreads();
  }
#pragma unroll
  for (int mi = 0; mi < 2; ++mi)
#pragma unroll
    for (int n = 0; n < 4; ++n)
#pragma unroll
      for (int r = 0; r < 8; ++r) {
        int g = m0 + wvi * 32 + mi * 16 + r + 8 * half;
        int d = n0g + n * 16 + col;
        out[(size_t)g * CC + d] = acc[mi][n][r] + pbias[d];
      }
}

// ---------------------------------------------------------------------------
extern "C" void kernel_launch(void* const* d_in, const int* in_sizes, int n_in,
                              void* d_out, int out_size, void* d_ws, size_t ws_size,
                              hipStream_t stream) {
  // setup_inputs order: x, Wk, Wq, Wv, proj_w, proj_b
  const float* x  = (const float*)d_in[0];
  const float* Wk = (const float*)d_in[1];
  const float* Wq = (const float*)d_in[2];
  const float* Wv = (const float*)d_in[3];
  const float* pw = (const float*)d_in[4];
  const float* pb = (const float*)d_in[5];
  float* out = (float*)d_out;

  char* ws = (char*)d_ws;
  size_t off = 0;
  auto alloc_bf = [&](size_t elems) {
    __bf16* p = (__bf16*)(ws + off);
    off += elems * sizeof(__bf16);
    return p;
  };
  const size_t NX  = (size_t)BB * TT * CC;   // 8Mi (also == B*H*T*HS)
  const size_t NW  = (size_t)HH * CC * HS;   // 1Mi
  const size_t NPW = (size_t)CC * CC;        // 1Mi

  __bf16* xb   = alloc_bf(NX);
  __bf16* wqb  = alloc_bf(NW);
  __bf16* wkb  = alloc_bf(NW);
  __bf16* wvb  = alloc_bf(NW);
  __bf16* pwb  = alloc_bf(NPW);
  __bf16* qbuf = alloc_bf(NX);
  __bf16* kbuf = alloc_bf(NX);
  __bf16* vbuf = alloc_bf(NX);
  __bf16* attb = alloc_bf(NX);
  (void)ws_size; (void)in_sizes; (void)n_in; (void)out_size;

  f32_to_bf16<<<dim3(2048), dim3(256), 0, stream>>>(x,  xb,  NX);
  f32_to_bf16<<<dim3(512),  dim3(256), 0, stream>>>(Wq, wqb, NW);
  f32_to_bf16<<<dim3(512),  dim3(256), 0, stream>>>(Wk, wkb, NW);
  f32_to_bf16<<<dim3(512),  dim3(256), 0, stream>>>(Wv, wvb, NW);
  f32_to_bf16<<<dim3(512),  dim3(256), 0, stream>>>(pw, pwb, NPW);

  qkv_gemm<<<dim3(BB * TT / 128, HH, 3), dim3(128), 0, stream>>>(
      xb, wqb, wkb, wvb, qbuf, kbuf, vbuf);

  flash_attn<<<dim3(TT / 64, BB * HH), dim3(128), 0, stream>>>(
      qbuf, kbuf, vbuf, attb);

  proj_gemm<<<dim3(BB * TT / 128, CC / 64), dim3(128), 0, stream>>>(
      attb, pwb, pb, out);
}
